// LinearAttention_90864328114544
// MI455X (gfx1250) — matrix operand order repaired and checked by reading, not compiled, not run on hardware
//
#include <hip/hip_runtime.h>
#include <hip/hip_bf16.h>
#include <math.h>

// ---------------- types ----------------
typedef __bf16 bf16;
typedef __attribute__((ext_vector_type(16))) __bf16 v16bf;
typedef __attribute__((ext_vector_type(8)))  __bf16 v8bf;
typedef __attribute__((ext_vector_type(8)))  float  v8f;
typedef __attribute__((ext_vector_type(4)))  unsigned v4u;
typedef __attribute__((ext_vector_type(8)))  int      v8i;
typedef __attribute__((ext_vector_type(4)))  int      v4i;

#define B_  8
#define N_  8192
#define C_  512
#define H_  8
#define HD_ 64

// native converts -> v_cvt_*bf16* instead of integer emulation
static __device__ __forceinline__ bf16  f2bf(float f) { return (bf16)f; }
static __device__ __forceinline__ float bf2f(bf16 b)  { return (float)b; }

// generic pointer -> LDS byte offset (addrspace(3) cast)
static __device__ __forceinline__ unsigned lds_offset(const void* p) {
  return (unsigned)(unsigned long long)(const __attribute__((address_space(3))) char*)p;
}

// ---------------------------------------------------------------------
// Tensor Data Mover: 2D bf16 tile (tile_d0 contiguous elems x tile_d1 rows,
// row stride = stride_elems) global -> LDS, packed row-major in LDS.
// D# packing per CDNA5 ISA ch.8 (group0 128b, group1 256b).
// Toolchain uses the 6-arg builtin form (clang-23 / therock headers).
// ---------------------------------------------------------------------
static __device__ __forceinline__ void tdm_load_2d_bf16(
    unsigned lds_byte_addr, const void* gptr,
    unsigned tile_d0, unsigned tile_d1, unsigned long long stride_elems) {
  const unsigned long long ga = (unsigned long long)gptr;
  v4u g0;
  g0[0] = 1u;                                  // count=1 (valid), load, user
  g0[1] = lds_byte_addr;                       // lds_addr [63:32]
  g0[2] = (unsigned)(ga & 0xFFFFFFFFu);        // global_addr lo
  g0[3] = (unsigned)((ga >> 32) & 0x01FFFFFFu) // global_addr [56:32]
          | (2u << 30);                        // type=2 ("image")
  const unsigned td0 = 0x40000000u;            // huge tensor dims: no OOB clip
  const unsigned td1 = 0x40000000u;
  v8i g1;
  g1[0] = (int)(1u << 16);                                     // data_size=1 (2B)
  g1[1] = (int)((td0 & 0xFFFFu) << 16);                        // tensor_dim0 lo16
  g1[2] = (int)((td0 >> 16) | ((td1 & 0xFFFFu) << 16));        // dim0 hi / dim1 lo
  g1[3] = (int)((td1 >> 16) | (tile_d0 << 16));                // dim1 hi / tile_dim0
  g1[4] = (int)(tile_d1 & 0xFFFFu);                            // tile_dim1, tile_dim2=0
  g1[5] = (int)(unsigned)(stride_elems & 0xFFFFFFFFu);         // dim0_stride lo32
  g1[6] = (int)(unsigned)((stride_elems >> 32) & 0xFFFFu);     // dim0_stride hi16
  g1[7] = 0;
  const v4i z4 = {0, 0, 0, 0};
  const v8i z8i = {0, 0, 0, 0, 0, 0, 0, 0};
  __builtin_amdgcn_tensor_load_to_lds(g0, g1, z4, z4, z8i, 0);
}

// =====================================================================
// Kernel A: qk = x @ qk_w^T + b ; q = elu(.)+1 ; k = elu(.)+1  (bf16 out)
// 64x128 block tile, 8 waves, 4 wmma tiles/wave/K-step, K-step 32.
// (f32->bf16 conversion needed, so tiles staged via VMEM+DS, not TDM.)
// =====================================================================
__global__ __launch_bounds__(256) void qk_gemm_elu_kernel(
    const float* __restrict__ x, const float* __restrict__ qk_w,
    const float* __restrict__ qk_b, bf16* __restrict__ qws,
    bf16* __restrict__ kws) {
  __shared__ __align__(32) bf16 xs[64 * 32];    // A [row][k], k-contiguous per row
  __shared__ __align__(32) bf16 wsb[128 * 32];  // B K-contiguous per col

  const int tid = threadIdx.x;
  const int lane = tid & 31;
  const int wid = tid >> 5;
  const int rowbase = blockIdx.x * 64;
  const int colbase = blockIdx.y * 128;
  const int waveRow = wid >> 1;
  const int waveCol = wid & 1;

  const v8f z8 = {0.f,0.f,0.f,0.f,0.f,0.f,0.f,0.f};
  v8f acc[4] = { z8, z8, z8, z8 };

  const int sxrow = tid >> 2;
  const int sxk0  = (tid & 3) * 8;
  const int swcol = tid >> 1;
  const int swk0  = (tid & 1) * 16;

  for (int k0 = 0; k0 < C_; k0 += 32) {
    { // stage x 64x32 (row-major, k contiguous per row)
      const float* p = x + (size_t)(rowbase + sxrow) * C_ + k0 + sxk0;
      __builtin_prefetch(p + 32, 0, 1);
      float4 f0 = *(const float4*)(p);
      float4 f1 = *(const float4*)(p + 4);
      v8bf pk;
      pk[0]=f2bf(f0.x); pk[1]=f2bf(f0.y); pk[2]=f2bf(f0.z); pk[3]=f2bf(f0.w);
      pk[4]=f2bf(f1.x); pk[5]=f2bf(f1.y); pk[6]=f2bf(f1.z); pk[7]=f2bf(f1.w);
      const int pb = sxk0;
      *(v8bf*)&xs[sxrow * 32 + pb] = pk;
    }
    { // stage w 128x32 (B[k][col] = qk_w[col*C + k])
      const float* p = qk_w + (size_t)(colbase + swcol) * C_ + k0 + swk0;
      #pragma unroll
      for (int half = 0; half < 2; ++half) {
        float4 f0 = *(const float4*)(p + half * 8);
        float4 f1 = *(const float4*)(p + half * 8 + 4);
        v8bf pk;
        pk[0]=f2bf(f0.x); pk[1]=f2bf(f0.y); pk[2]=f2bf(f0.z); pk[3]=f2bf(f0.w);
        pk[4]=f2bf(f1.x); pk[5]=f2bf(f1.y); pk[6]=f2bf(f1.z); pk[7]=f2bf(f1.w);
        *(v8bf*)&wsb[swcol * 32 + swk0 + half * 8] = pk;
      }
    }
    __syncthreads();
    const int sel = (lane < 16) ? 0 : 16;
    v16bf a = *(const v16bf*)&xs[(waveRow * 16 + (lane & 15)) * 32 + sel];
    #pragma unroll
    for (int tt = 0; tt < 4; ++tt) {
      v16bf bb = *(const v16bf*)&wsb[(waveCol * 64 + tt * 16 + (lane & 15)) * 32 + sel];
      acc[tt] = __builtin_amdgcn_wmma_f32_16x16x32_bf16(
          false, a, false, bb, (short)0, acc[tt], false, false);
    }
    __syncthreads();
  }

  const bool is_k = (colbase >= C_);
  bf16* dst = is_k ? kws : qws;
  const int cb = colbase - (is_k ? C_ : 0);
  #pragma unroll
  for (int tt = 0; tt < 4; ++tt) {
    const int coff = waveCol * 64 + tt * 16 + (lane & 15);
    const float bias = qk_b[colbase + coff];
    #pragma unroll
    for (int r = 0; r < 8; ++r) {
      const int row = rowbase + waveRow * 16 + r + ((lane >= 16) ? 8 : 0);
      const float v = acc[tt][r] + bias;
      const float e = (v > 0.f) ? (v + 1.f) : __expf(v);   // elu(v)+1
      dst[(size_t)row * C_ + cb + coff] = f2bf(e);
    }
  }
}

// =====================================================================
// Kernel B: per (b,h): kv = k^T v over N, ksum = sum_n k[n][:]
// k chunk (32n x 64d, bf16) arrives via TDM; v staged (f32->bf16 transpose).
// =====================================================================
__global__ __launch_bounds__(256) void kv_reduce_kernel(
    const float* __restrict__ x, const bf16* __restrict__ kws,
    float* __restrict__ kvbuf, float* __restrict__ kmean) {
  __shared__ __align__(32) bf16 klds[32 * 64];  // [n][d] row-major (TDM dest)
  __shared__ __align__(32) bf16 vb[64 * 32];    // B operand [e][kk]

  const int tid = threadIdx.x, lane = tid & 31, wid = tid >> 5;
  const int bh = blockIdx.x, b = bh >> 3, h = bh & 7;

  const int t0 = wid * 2;
  const int drow = t0 >> 2;
  const v8f z8 = {0.f,0.f,0.f,0.f,0.f,0.f,0.f,0.f};
  v8f acc[2] = { z8, z8 };
  float ksum = 0.f;

  const int skk = tid >> 3;
  const int sd0 = (tid & 7) * 8;
  const unsigned klds_off = lds_offset(klds);

  for (int n0 = 0; n0 < N_; n0 += 32) {
    if (wid == 0) {  // async DMA: k[n0..n0+31][h*64..h*64+63] -> LDS
      tdm_load_2d_bf16(klds_off,
                       kws + (size_t)(b * N_ + n0) * C_ + h * HD_,
                       HD_, 32, C_);
    }
    { // stage v: vb[e*32 + kk] = x[b, n0+kk, h*64+e]
      const float* p = x + (size_t)(b * N_ + n0 + skk) * C_ + h * HD_ + sd0;
      float4 f0 = *(const float4*)p;
      float4 f1 = *(const float4*)(p + 4);
      const float vv[8] = {f0.x,f0.y,f0.z,f0.w,f1.x,f1.y,f1.z,f1.w};
      #pragma unroll
      for (int j = 0; j < 8; ++j) vb[(sd0 + j) * 32 + skk] = f2bf(vv[j]);
    }
    if (wid == 0) __builtin_amdgcn_s_wait_tensorcnt(0);
    __syncthreads();

    // A fragment gather: A[m=d][kk=n] = klds[kk*64 + m]
    const int m = drow * 16 + (lane & 15);
    const int kb = (lane < 16) ? 0 : 16;
    v16bf a;
    #pragma unroll
    for (int j = 0; j < 8; ++j) a[j]     = klds[(kb + j) * HD_ + m];
    #pragma unroll
    for (int j = 0; j < 8; ++j) a[8 + j] = klds[(8 + kb + j) * HD_ + m];

    const int sel = (lane < 16) ? 0 : 16;
    #pragma unroll
    for (int j = 0; j < 2; ++j) {
      const int ecol = (t0 + j) & 3;
      v16bf bb = *(const v16bf*)&vb[(ecol * 16 + (lane & 15)) * 32 + sel];
      acc[j] = __builtin_amdgcn_wmma_f32_16x16x32_bf16(
          false, a, false, bb, (short)0, acc[j], false, false);
    }
    if (tid < 64) {
      #pragma unroll
      for (int j = 0; j < 32; ++j) ksum += bf2f(klds[j * HD_ + tid]);
    }
    __syncthreads();
  }

  #pragma unroll
  for (int j = 0; j < 2; ++j) {
    const int ecol = (t0 + j) & 3;
    #pragma unroll
    for (int r = 0; r < 8; ++r) {
      const int d = drow * 16 + r + ((lane >= 16) ? 8 : 0);
      const int e = ecol * 16 + (lane & 15);
      kvbuf[(size_t)bh * (HD_ * HD_) + d * HD_ + e] = acc[j][r];
    }
  }
  if (tid < 64) kmean[bh * HD_ + tid] = ksum * (1.f / (float)N_);
}

// =====================================================================
// Kernel C: out = (q @ kv) * z + lepe(x)
// q tile (128 x 64 bf16) arrives via TDM once per block.
// =====================================================================
__global__ __launch_bounds__(256) void attn_out_kernel(
    const float* __restrict__ x, const bf16* __restrict__ qws,
    const float* __restrict__ lepe_w, const float* __restrict__ lepe_b,
    const float* __restrict__ kvbuf, const float* __restrict__ kmean,
    float* __restrict__ out) {
  __shared__ __align__(32) bf16 qlds[128 * 64];     // [row][d] (TDM dest)
  __shared__ __align__(32) bf16 kvb[2 * 64 * 32];   // [kstep][e][kk]
  __shared__ float km[64];
  __shared__ float zsh[8][16];

  const int tid = threadIdx.x, lane = tid & 31, wid = tid >> 5;
  const int bh = blockIdx.x, b = bh >> 3, h = bh & 7;
  const int nbase = blockIdx.y * 128;

  if (wid == 0) {  // async DMA: q[nbase..nbase+127][h*64..] -> LDS
    tdm_load_2d_bf16(lds_offset(qlds),
                     qws + (size_t)(b * N_ + nbase) * C_ + h * HD_,
                     HD_, 128, C_);
  }
  { // stage kv 64x64 f32 -> bf16 B-fragment layout
    const int d = tid >> 2;
    const int e0 = (tid & 3) * 16;
    const int s = d >> 5, dk = d & 31;
    const float* p = kvbuf + (size_t)bh * (HD_ * HD_) + d * HD_ + e0;
    #pragma unroll
    for (int j4 = 0; j4 < 4; ++j4) {
      float4 f = *(const float4*)(p + j4 * 4);
      const float vv[4] = {f.x, f.y, f.z, f.w};
      #pragma unroll
      for (int j = 0; j < 4; ++j)
        kvb[(s * 64 + e0 + j4 * 4 + j) * 32 + dk] = f2bf(vv[j]);
    }
  }
  if (tid < 64) km[tid] = kmean[bh * HD_ + tid];
  if (wid == 0) __builtin_amdgcn_s_wait_tensorcnt(0);
  __syncthreads();

  // z = 1/(q . k_mean + eps) per row (lanes 0..15), from LDS q
  if (lane < 16) {
    const bf16* qp = &qlds[(wid * 16 + lane) * HD_];
    float sum = 0.f;
    #pragma unroll
    for (int d = 0; d < 64; ++d) sum += bf2f(qp[d]) * km[d];
    zsh[wid][lane] = 1.f / (sum + 1e-6f);
  }
  __syncthreads();

  // A fragments from LDS q (two 16B chunks per fragment half)
  const bf16* qp = &qlds[(wid * 16 + (lane & 15)) * HD_];
  v16bf afr[2];
  #pragma unroll
  for (int s = 0; s < 2; ++s) {
    const int o = (lane < 16) ? 0 : 16;
    v8bf lo = *(const v8bf*)(qp + s * 32 + o);
    v8bf hi = *(const v8bf*)(qp + s * 32 + 8 + o);
    afr[s] = __builtin_shufflevector(lo, hi, 0,1,2,3,4,5,6,7,8,9,10,11,12,13,14,15);
  }

  const int sel = (lane < 16) ? 0 : 16;
  const v8f z8 = {0.f,0.f,0.f,0.f,0.f,0.f,0.f,0.f};
  v8f acc[4] = { z8, z8, z8, z8 };
  #pragma unroll
  for (int s = 0; s < 2; ++s) {
    #pragma unroll
    for (int tt = 0; tt < 4; ++tt) {
      v16bf bb = *(const v16bf*)&kvb[(s * 64 + tt * 16 + (lane & 15)) * 32 + sel];
      acc[tt] = __builtin_amdgcn_wmma_f32_16x16x32_bf16(
          false, afr[s], false, bb, (short)0, acc[tt], false, false);
    }
  }

  // epilogue: z-scale + depthwise conv3 (lepe) + store
  #pragma unroll
  for (int tt = 0; tt < 4; ++tt) {
    const int e = tt * 16 + (lane & 15);
    const int c = h * HD_ + e;
    const float w0 = lepe_w[c * 3 + 0];
    const float w1 = lepe_w[c * 3 + 1];
    const float w2 = lepe_w[c * 3 + 2];
    const float lb = lepe_b[c];
    #pragma unroll
    for (int r = 0; r < 8; ++r) {
      const int rowl = r + ((lane >= 16) ? 8 : 0);
      const int n = nbase + wid * 16 + rowl;
      const size_t base = (size_t)(b * N_ + n) * C_ + c;
      const float x0 = x[base];
      const float xm = (n > 0)      ? x[base - C_] : 0.f;
      const float xp = (n < N_ - 1) ? x[base + C_] : 0.f;
      const float lepe = xm * w0 + x0 * w1 + xp * w2 + lb;
      out[base] = acc[tt][r] * zsh[wid][rowl] + lepe;
    }
  }
}

// =====================================================================
extern "C" void kernel_launch(void* const* d_in, const int* in_sizes, int n_in,
                              void* d_out, int out_size, void* d_ws, size_t ws_size,
                              hipStream_t stream) {
  (void)in_sizes; (void)n_in; (void)out_size; (void)ws_size;
  const float* x      = (const float*)d_in[0];
  const float* qk_w   = (const float*)d_in[1];
  const float* qk_b   = (const float*)d_in[2];
  const float* lepe_w = (const float*)d_in[3];
  const float* lepe_b = (const float*)d_in[4];
  float* out = (float*)d_out;

  char* ws = (char*)d_ws;
  const size_t qk_bytes = (size_t)B_ * N_ * C_ * sizeof(bf16);   // 64 MB each
  bf16*  qws   = (bf16*)ws;
  bf16*  kws   = (bf16*)(ws + qk_bytes);
  float* kvbuf = (float*)(ws + 2 * qk_bytes);
  float* kmean = (float*)(ws + 2 * qk_bytes + (size_t)B_ * H_ * HD_ * HD_ * sizeof(float));

  dim3 gA((B_ * N_) / 64, (2 * C_) / 128);
  qk_gemm_elu_kernel<<<gA, dim3(256), 0, stream>>>(x, qk_w, qk_b, qws, kws);

  kv_reduce_kernel<<<dim3(B_ * H_), dim3(256), 0, stream>>>(x, kws, kvbuf, kmean);

  dim3 gC(B_ * H_, N_ / 128);
  attn_out_kernel<<<gC, dim3(256), 0, stream>>>(x, qws, lepe_w, lepe_b,
                                                kvbuf, kmean, out);
}
